// MultiHeadAttention_44332652429419
// MI455X (gfx1250) — matrix operand order repaired and checked by reading, not compiled, not run on hardware
//
// CDNA5 / gfx1250 fused graph multi-head attention.
// bf16 WMMA (v_wmma_f32_16x16x32_bf16) for all GEMMs + flash attention.
#include <hip/hip_runtime.h>

#define NN    3072
#define INF_  1024
#define OUTF  1024
#define NH    16
#define HD    64
#define NET   4

typedef __attribute__((ext_vector_type(16))) __bf16 v16bf;
typedef __attribute__((ext_vector_type(8)))  float  v8f;

union FragU { unsigned int u[8]; v16bf v; };

static __device__ __forceinline__ unsigned short f2bf(float f) {
    unsigned int u = __float_as_uint(f);
    u += 0x7FFFu + ((u >> 16) & 1u);          // round-to-nearest-even
    return (unsigned short)(u >> 16);
}

static __device__ __forceinline__ v8f wmma_bf16(const FragU& a, const FragU& b, v8f c) {
    return __builtin_amdgcn_wmma_f32_16x16x32_bf16(false, a.v, false, b.v,
                                                   (short)0, c, false, false);
}

// ---------------- elementwise conversion kernels ----------------
__global__ void k_f2bf(const float* __restrict__ x, unsigned short* __restrict__ y, int n) {
    int i = blockIdx.x * blockDim.x + threadIdx.x;
    if (i < n) y[i] = f2bf(x[i]);
}

// wt[i*R + o] = bf16(w[o*C + i])  (transpose + convert, R==C==1024)
__global__ void k_transpose_bf16(const float* __restrict__ w, unsigned short* __restrict__ wt,
                                 int R, int C) {
    int idx = blockIdx.x * blockDim.x + threadIdx.x;
    if (idx < R * C) {
        int o = idx / C, i = idx % C;
        wt[(size_t)i * R + o] = f2bf(w[idx]);
    }
}

// bc[o] = sum_k wi[o][k]*bin[k] + badd[o]
__global__ void k_bias_combine(const float* __restrict__ wi, const float* __restrict__ bin,
                               const float* __restrict__ badd, float* __restrict__ bc, int n) {
    int o = blockIdx.x * blockDim.x + threadIdx.x;
    if (o < n) {
        float s = badd[o];
        const float* r = wi + (size_t)o * n;
        for (int k = 0; k < n; ++k) s += r[k] * bin[k];
        bc[o] = s;
    }
}

__global__ void k_softplus(const float* __restrict__ e, float* __restrict__ w) {
    int i = threadIdx.x;
    if (i < NET) w[i] = log1pf(__expf(e[i]));
}

// ---------------- mask build ----------------
__global__ void k_mask_init(float* __restrict__ mask) {
    long i = (long)blockIdx.x * blockDim.x + threadIdx.x;
    if (i < (long)NN * NN) mask[i] = -INFINITY;
}
__global__ void k_mask_scatter(const int* __restrict__ ei, const int* __restrict__ et,
                               const float* __restrict__ w, float* __restrict__ mask, int E) {
    int e = blockIdx.x * blockDim.x + threadIdx.x;
    if (e < E) {
        int s = ei[e], d = ei[E + e];
        mask[(size_t)s * NN + d] = w[et[e] - 1];
    }
}
__global__ void k_mask_diag(const float* __restrict__ w, float* __restrict__ mask) {
    int n = blockIdx.x * blockDim.x + threadIdx.x;
    if (n < NN) {
        size_t p = (size_t)n * NN + n;
        if (mask[p] == -INFINITY) mask[p] = w[3];
    }
}

// ---------------- generic bf16 WMMA GEMM:  C[M,Nn] = A[M,Kk] @ Bw[Nn,Kk]^T + bias ----
// Each wave computes a 32(M) x 64(N) tile: 8 C-fragments, K stepped by 32.
enum { EPI_F32 = 0, EPI_BF16 = 1, EPI_BF16_SCALE = 2, EPI_VT = 3 };

template<int EPI>
__global__ __launch_bounds__(256) void k_gemm(const unsigned int* __restrict__ A,
                                              const unsigned int* __restrict__ Bw,
                                              const float* __restrict__ bias,
                                              void* __restrict__ Cout,
                                              int M, int Nn, int Kk, float scale) {
    const int lane = threadIdx.x & 31;
    const int half = lane >> 4;
    const int nlo  = lane & 15;
    const int wave = (blockIdx.x * blockDim.x + threadIdx.x) >> 5;
    const int tN = Nn >> 6;
    const int m0 = (wave / tN) << 5;
    const int n0 = (wave % tN) << 6;
    if (m0 >= M) return;
    const int K2u = Kk >> 1;                 // row stride in dwords (bf16 pairs)

    // A-matrix 16x32 bf16 contiguous k order: k = 16*half + s (dword indices within a 16-dword K-block)
    int ku[8];
#pragma unroll
    for (int v = 0; v < 8; ++v) ku[v] = v + 8 * half;

    v8f acc[2][4];
#pragma unroll
    for (int rb = 0; rb < 2; ++rb)
#pragma unroll
        for (int c = 0; c < 4; ++c) acc[rb][c] = {};

    for (int kb = 0; kb < Kk; kb += 32) {
        FragU a[2];
#pragma unroll
        for (int rb = 0; rb < 2; ++rb) {
            const unsigned int* ar = A + (size_t)(m0 + rb * 16 + nlo) * K2u + (kb >> 1);
#pragma unroll
            for (int v = 0; v < 8; ++v) a[rb].u[v] = ar[ku[v]];
        }
#pragma unroll
        for (int c = 0; c < 4; ++c) {
            const unsigned int* br =
                Bw + (size_t)(n0 + c * 16 + nlo) * K2u + (kb >> 1) + 8 * half;
            FragU b;
            *(uint4*)&b.u[0] = *(const uint4*)br;        // one contiguous 32B B-frag load
            *(uint4*)&b.u[4] = *(const uint4*)(br + 4);
            acc[0][c] = wmma_bf16(a[0], b, acc[0][c]);
            acc[1][c] = wmma_bf16(a[1], b, acc[1][c]);
        }
    }

#pragma unroll
    for (int rb = 0; rb < 2; ++rb)
#pragma unroll
        for (int c = 0; c < 4; ++c) {
            int col = n0 + c * 16 + nlo;
            float bv = bias ? bias[col] : 0.0f;
#pragma unroll
            for (int r = 0; r < 8; ++r) {
                int row = m0 + rb * 16 + r + 8 * half;   // C-frag: m = r + 8*(lane>>4)
                float val = acc[rb][c][r] + bv;
                if (EPI == EPI_BF16_SCALE) val *= scale;
                if (EPI == EPI_F32) {
                    ((float*)Cout)[(size_t)row * Nn + col] = val;
                } else if (EPI == EPI_VT) {
                    // per-head transposed store: [h*64+d][N] == [col][M]
                    ((unsigned short*)Cout)[(size_t)col * M + row] = f2bf(val);
                } else {
                    ((unsigned short*)Cout)[(size_t)row * Nn + col] = f2bf(val);
                }
            }
        }
}

// ---------------- flash attention: one wave per (head, 16-row query stripe) ---------
__global__ __launch_bounds__(256) void k_attn(const unsigned int* __restrict__ q2,
                                              const unsigned int* __restrict__ k2,
                                              const unsigned int* __restrict__ vt,
                                              const float* __restrict__ mask,
                                              unsigned short* __restrict__ attn) {
    __shared__ unsigned short pbuf[8][16][32];   // per-wave P staging (C-layout -> A-layout)
    const int lane = threadIdx.x & 31, half = lane >> 4, nlo = lane & 15;
    const int wv = threadIdx.x >> 5;
    const int gw = blockIdx.x * 8 + wv;          // 16 heads * 192 stripes = 3072 waves
    const int h  = gw / (NN / 16);
    const int q0 = (gw % (NN / 16)) * 16;

    int ku[8];
#pragma unroll
    for (int v = 0; v < 8; ++v) ku[v] = v + 8 * half;

    // Q A-fragments (scale 1/sqrt(D) already folded in at projection time)
    FragU aq[2];
    const unsigned int* qr = q2 + (size_t)(q0 + nlo) * (OUTF / 2) + h * (HD / 2);
#pragma unroll
    for (int d = 0; d < 2; ++d)
#pragma unroll
        for (int v = 0; v < 8; ++v) aq[d].u[v] = qr[d * 16 + ku[v]];

    v8f o[4];
#pragma unroll
    for (int c = 0; c < 4; ++c) o[c] = {};
    float mrow[8], lrow[8];
#pragma unroll
    for (int r = 0; r < 8; ++r) { mrow[r] = -1e30f; lrow[r] = 0.0f; }

    for (int j = 0; j < NN; j += 32) {
        // S = Q K^T over 32 keys (two 16x16 fragments), contraction D=64 in two steps
        v8f s0 = {}, s1 = {};
#pragma unroll
        for (int d = 0; d < 2; ++d) {
            const unsigned int* kr =
                k2 + (size_t)(j + nlo) * (OUTF / 2) + h * (HD / 2) + d * 16 + 8 * half;
            FragU b0, b1;
            *(uint4*)&b0.u[0] = *(const uint4*)kr;
            *(uint4*)&b0.u[4] = *(const uint4*)(kr + 4);
            const unsigned int* kr1 = kr + 16 * (OUTF / 2);
            *(uint4*)&b1.u[0] = *(const uint4*)kr1;
            *(uint4*)&b1.u[4] = *(const uint4*)(kr1 + 4);
            s0 = wmma_bf16(aq[d], b0, s0);
            s1 = wmma_bf16(aq[d], b1, s1);
        }

        // mask add + online softmax (row stats over 16 lanes of each half-wave)
        float corr[8];
#pragma unroll
        for (int r = 0; r < 8; ++r) {
            const float* mrp = mask + (size_t)(q0 + r + 8 * half) * NN + j + nlo;
            float a = s0[r] + mrp[0];
            float b = s1[r] + mrp[16];
            float t = fmaxf(a, b);
#pragma unroll
            for (int off = 1; off < 16; off <<= 1) t = fmaxf(t, __shfl_xor(t, off, 32));
            float mn = fmaxf(mrow[r], t);
            corr[r] = __expf(mrow[r] - mn);
            float e0 = __expf(a - mn), e1 = __expf(b - mn);
            float sm = e0 + e1;
#pragma unroll
            for (int off = 1; off < 16; off <<= 1) sm += __shfl_xor(sm, off, 32);
            lrow[r] = lrow[r] * corr[r] + sm;
            mrow[r] = mn;
            pbuf[wv][r + 8 * half][nlo]      = f2bf(e0);
            pbuf[wv][r + 8 * half][16 + nlo] = f2bf(e1);
        }
#pragma unroll
        for (int c = 0; c < 4; ++c)
#pragma unroll
            for (int r = 0; r < 8; ++r) o[c][r] *= corr[r];

        // read P back in A-fragment order (LDS ops in-order within a wave)
        FragU pa;
        const unsigned int* pr = (const unsigned int*)&pbuf[wv][nlo][0];
#pragma unroll
        for (int v = 0; v < 8; ++v) pa.u[v] = pr[ku[v]];

        // O += P V : V stored transposed [h*64+d][N] -> contiguous 32B B-frag loads
#pragma unroll
        for (int c = 0; c < 4; ++c) {
            const unsigned int* vr =
                vt + (size_t)(h * HD + c * 16 + nlo) * (NN / 2) + (j >> 1) + 8 * half;
            FragU b;
            *(uint4*)&b.u[0] = *(const uint4*)vr;
            *(uint4*)&b.u[4] = *(const uint4*)(vr + 4);
            o[c] = wmma_bf16(pa, b, o[c]);
        }
    }

#pragma unroll
    for (int c = 0; c < 4; ++c)
#pragma unroll
        for (int r = 0; r < 8; ++r) {
            float val = o[c][r] / lrow[r];
            attn[(size_t)(q0 + r + 8 * half) * OUTF + h * HD + c * 16 + nlo] = f2bf(val);
        }
}

// ---------------- host orchestration ----------------
extern "C" void kernel_launch(void* const* d_in, const int* in_sizes, int n_in,
                              void* d_out, int out_size, void* d_ws, size_t ws_size,
                              hipStream_t stream) {
    (void)n_in; (void)out_size; (void)ws_size;
    const float* x   = (const float*)d_in[0];
    const int*   ei  = (const int*)d_in[1];
    const int*   et  = (const int*)d_in[2];
    const float* etw = (const float*)d_in[3];
    const float* wq  = (const float*)d_in[4];  const float* bq  = (const float*)d_in[5];
    const float* wk  = (const float*)d_in[6];  const float* bk  = (const float*)d_in[7];
    const float* wvv = (const float*)d_in[8];  const float* bv  = (const float*)d_in[9];
    const float* wiq = (const float*)d_in[10]; const float* biq = (const float*)d_in[11];
    const float* wik = (const float*)d_in[12]; const float* bik = (const float*)d_in[13];
    const float* wiv = (const float*)d_in[14]; const float* biv = (const float*)d_in[15];
    const float* wo  = (const float*)d_in[16]; const float* bo  = (const float*)d_in[17];
    float* out = (float*)d_out;
    const int E = in_sizes[2];

    char* p = (char*)d_ws;
    auto carve = [&](size_t bytes) -> void* {
        void* r = (void*)p;
        p += (bytes + 255) & ~(size_t)255;
        return r;
    };
    unsigned short* xb   = (unsigned short*)carve((size_t)NN * INF_ * 2);
    unsigned short* wqT  = (unsigned short*)carve((size_t)1024 * 1024 * 2);
    unsigned short* wkT  = (unsigned short*)carve((size_t)1024 * 1024 * 2);
    unsigned short* wvT  = (unsigned short*)carve((size_t)1024 * 1024 * 2);
    unsigned short* wiqB = (unsigned short*)carve((size_t)1024 * 1024 * 2);
    unsigned short* wikB = (unsigned short*)carve((size_t)1024 * 1024 * 2);
    unsigned short* wivB = (unsigned short*)carve((size_t)1024 * 1024 * 2);
    unsigned short* woB  = (unsigned short*)carve((size_t)1024 * 1024 * 2);
    unsigned short* WcQ  = (unsigned short*)carve((size_t)1024 * 1024 * 2);
    unsigned short* WcK  = (unsigned short*)carve((size_t)1024 * 1024 * 2);
    unsigned short* WcV  = (unsigned short*)carve((size_t)1024 * 1024 * 2);
    float* bcQ = (float*)carve(1024 * 4);
    float* bcK = (float*)carve(1024 * 4);
    float* bcV = (float*)carve(1024 * 4);
    float* spw = (float*)carve(256);
    unsigned short* q2b  = (unsigned short*)carve((size_t)NN * OUTF * 2);
    unsigned short* k2b  = (unsigned short*)carve((size_t)NN * OUTF * 2);
    unsigned short* vtb  = (unsigned short*)carve((size_t)NN * OUTF * 2);
    unsigned short* atnb = (unsigned short*)carve((size_t)NN * OUTF * 2);
    float* mask = (float*)carve((size_t)NN * NN * 4);

    // 1) conversions
    k_f2bf<<<(NN * INF_) / 256, 256, 0, stream>>>(x, xb, NN * INF_);
    k_transpose_bf16<<<4096, 256, 0, stream>>>(wq,  wqT, 1024, 1024);
    k_transpose_bf16<<<4096, 256, 0, stream>>>(wk,  wkT, 1024, 1024);
    k_transpose_bf16<<<4096, 256, 0, stream>>>(wvv, wvT, 1024, 1024);
    k_f2bf<<<4096, 256, 0, stream>>>(wiq, wiqB, 1024 * 1024);
    k_f2bf<<<4096, 256, 0, stream>>>(wik, wikB, 1024 * 1024);
    k_f2bf<<<4096, 256, 0, stream>>>(wiv, wivB, 1024 * 1024);
    k_f2bf<<<4096, 256, 0, stream>>>(wo,  woB,  1024 * 1024);

    // 2) fold projection pairs: Wc = wi @ w ; bc = wi @ b + bi
    k_gemm<EPI_BF16><<<64, 256, 0, stream>>>((const unsigned int*)wiqB, (const unsigned int*)wqT,
                                             nullptr, WcQ, 1024, 1024, 1024, 1.0f);
    k_gemm<EPI_BF16><<<64, 256, 0, stream>>>((const unsigned int*)wikB, (const unsigned int*)wkT,
                                             nullptr, WcK, 1024, 1024, 1024, 1.0f);
    k_gemm<EPI_BF16><<<64, 256, 0, stream>>>((const unsigned int*)wivB, (const unsigned int*)wvT,
                                             nullptr, WcV, 1024, 1024, 1024, 1.0f);
    k_bias_combine<<<4, 256, 0, stream>>>(wiq, bq, biq, bcQ, 1024);
    k_bias_combine<<<4, 256, 0, stream>>>(wik, bk, bik, bcK, 1024);
    k_bias_combine<<<4, 256, 0, stream>>>(wiv, bv, biv, bcV, 1024);

    // 3) fused projections (Q scaled by 1/sqrt(D)=0.125; V stored per-head transposed)
    k_gemm<EPI_BF16_SCALE><<<192, 256, 0, stream>>>((const unsigned int*)xb,
        (const unsigned int*)WcQ, bcQ, q2b, NN, OUTF, INF_, 0.125f);
    k_gemm<EPI_BF16><<<192, 256, 0, stream>>>((const unsigned int*)xb,
        (const unsigned int*)WcK, bcK, k2b, NN, OUTF, INF_, 1.0f);
    k_gemm<EPI_VT><<<192, 256, 0, stream>>>((const unsigned int*)xb,
        (const unsigned int*)WcV, bcV, vtb, NN, OUTF, INF_, 1.0f);

    // 4) mask
    k_softplus<<<1, 32, 0, stream>>>(etw, spw);
    k_mask_init<<<(int)(((long)NN * NN) / 256), 256, 0, stream>>>(mask);
    k_mask_scatter<<<(E + 255) / 256, 256, 0, stream>>>(ei, et, spw, mask, E);
    k_mask_diag<<<(NN + 255) / 256, 256, 0, stream>>>(spw, mask);

    // 5) flash attention (3072 waves, 8 per block)
    k_attn<<<NN * NH / 16 / 8, 256, 0, stream>>>((const unsigned int*)q2b,
        (const unsigned int*)k2b, (const unsigned int*)vtb, mask, atnb);

    // 6) out projection -> fp32 output
    k_gemm<EPI_F32><<<192, 256, 0, stream>>>((const unsigned int*)atnb,
        (const unsigned int*)woB, bo, out, NN, OUTF, OUTF, 1.0f);
}
